// GCN_58720792871132
// MI455X (gfx1250) — matrix operand order repaired and checked by reading, not compiled, not run on hardware
//
#include <hip/hip_runtime.h>
#include <math.h>

typedef __attribute__((ext_vector_type(16))) _Float16 v16h;
typedef __attribute__((ext_vector_type(8)))  _Float16 v8h;
typedef __attribute__((ext_vector_type(4)))  _Float16 v4h;
typedef __attribute__((ext_vector_type(8)))  float    v8f;

#define N_NODES   100000
#define N_EDGES   1600000
#define HIDDEN    128
#define N_GRAPHS  64
#define N_CLASSES 10

// ---------------------------------------------------------------- degree + scalar agg (layer 1 input is d=1)
__global__ void deg_agg1_kernel(const int* __restrict__ src, const int* __restrict__ dst,
                                const float* __restrict__ x, float* __restrict__ deg,
                                float* __restrict__ agg1) {
  int e = blockIdx.x * 256 + threadIdx.x;
  if (e < N_EDGES) {
    int d = dst[e];
    atomicAdd(&deg[d], 1.0f);
    atomicAdd(&agg1[d], x[src[e]]);
  }
}

// ---------------------------------------------------------------- layer 1: rank-1, elementwise
__global__ void layer1_kernel(const float* __restrict__ x, const float* __restrict__ agg1,
                              const float* __restrict__ deg, const float* __restrict__ W1l,
                              const float* __restrict__ W1r, const float* __restrict__ b1,
                              float* __restrict__ out) {
  int tid = blockIdx.x * 256 + threadIdx.x;   // exactly N_NODES*128 threads
  int n = tid >> 7, j = tid & 127;
  float mean = agg1[n] / fmaxf(deg[n], 1.0f);
  float v = mean * W1l[j] + b1[j] + x[n] * W1r[j];
  out[tid] = fmaxf(v, 0.0f);
}

// ---------------------------------------------------------------- edge scatter-add of 128-wide rows (L2 atomics)
__global__ void scatter128_kernel(const float* __restrict__ h, const int* __restrict__ src,
                                  const int* __restrict__ dst, float* __restrict__ agg) {
  int tid = blockIdx.x * 256 + threadIdx.x;   // one wave per edge, lane covers 4 floats
  int e = tid >> 5, lane = tid & 31;
  int s = src[e], d = dst[e];
  const float4 v = ((const float4*)h)[(size_t)s * 32 + lane];
  float* ap = agg + (size_t)d * 128 + lane * 4;
  atomicAdd(ap + 0, v.x);
  atomicAdd(ap + 1, v.y);
  atomicAdd(ap + 2, v.z);
  atomicAdd(ap + 3, v.w);
}

// ---------------------------------------------------------------- pack W[128,128] f32 -> WMMA-B-fragment-ordered f16
// packed[((kk*8+n)*32+lane)*16 + e] = (f16) W[K*128 + N],  K = 32kk + (lane<16?0:8) + (e&7) + (e>>3)*16,  N = 16n + (lane&15)
__global__ void pack_w_kernel(const float* __restrict__ W, _Float16* __restrict__ out) {
  int t = blockIdx.x * 256 + threadIdx.x;     // 16384 threads
  if (t < 16384) {
    int e = t & 15, lane = (t >> 4) & 31, n = (t >> 9) & 7, kk = t >> 12;
    int K = 32 * kk + ((lane >> 4) & 1) * 8 + (e & 7) + (e >> 3) * 16;
    int N = 16 * n + (lane & 15);
    out[t] = (_Float16)W[K * 128 + N];
  }
}

// ---------------------------------------------------------------- fused SAGE layer: relu((agg/deg)@Wl + h@Wr + b)
// one wave per 16-row tile; 8 v_wmma_f32_16x16x32_f16 per 16x16 output tile
__launch_bounds__(32)
__global__ void sage_gemm_kernel(const float* __restrict__ agg, const float* __restrict__ deg,
                                 const float* __restrict__ h, const _Float16* __restrict__ wl,
                                 const _Float16* __restrict__ wr, const float* __restrict__ bias,
                                 float* __restrict__ out) {
  __shared__ __align__(16) _Float16 ldsA[16 * 136];   // 16B row pad: bank-rotated, 16B-aligned chunks
  __shared__ __align__(16) _Float16 ldsH[16 * 136];
  const int lane = threadIdx.x;
  const int r0 = blockIdx.x * 16;

  const float4* aggV = (const float4*)agg;
  const float4* hV   = (const float4*)h;
#pragma unroll
  for (int row = 0; row < 16; ++row) {
    int gr = r0 + row;
    float rd = 1.0f / fmaxf(deg[gr], 1.0f);
    float4 va = aggV[(size_t)gr * 32 + lane];
    float4 vh = hV[(size_t)gr * 32 + lane];
    v4h pa = { (_Float16)(va.x * rd), (_Float16)(va.y * rd),
               (_Float16)(va.z * rd), (_Float16)(va.w * rd) };
    v4h ph = { (_Float16)vh.x, (_Float16)vh.y, (_Float16)vh.z, (_Float16)vh.w };
    *(v4h*)(ldsA + row * 136 + lane * 4) = pa;
    *(v4h*)(ldsH + row * 136 + lane * 4) = ph;
  }
  asm volatile("s_wait_dscnt 0" ::: "memory");   // single-wave LDS RAW fence

  const int rsel = lane & 15;             // A row / C col within tile
  const int hi   = (lane >> 4) & 1;       // lane half selector
  v16h fa[4], fh[4];                      // A fragments: mean (agg/deg) and self (h), K = 4 x 32
#pragma unroll
  for (int kk = 0; kk < 4; ++kk) {
    int base = 32 * kk + hi * 8;
    v8h lo = *(const v8h*)(ldsA + rsel * 136 + base);
    v8h hg = *(const v8h*)(ldsA + rsel * 136 + base + 16);
#pragma unroll
    for (int i = 0; i < 8; ++i) { fa[kk][i] = lo[i]; fa[kk][i + 8] = hg[i]; }
    lo = *(const v8h*)(ldsH + rsel * 136 + base);
    hg = *(const v8h*)(ldsH + rsel * 136 + base + 16);
#pragma unroll
    for (int i = 0; i < 8; ++i) { fh[kk][i] = lo[i]; fh[kk][i + 8] = hg[i]; }
  }

#pragma unroll
  for (int n = 0; n < 8; ++n) {
    v8f acc = {};
#pragma unroll
    for (int kk = 0; kk < 4; ++kk) {
      v16h bw = *(const v16h*)(wl + ((size_t)(kk * 8 + n) * 32 + lane) * 16);
      acc = __builtin_amdgcn_wmma_f32_16x16x32_f16(false, fa[kk], false, bw,
                                                   (short)0, acc, false, false);
    }
#pragma unroll
    for (int kk = 0; kk < 4; ++kk) {
      v16h bw = *(const v16h*)(wr + ((size_t)(kk * 8 + n) * 32 + lane) * 16);
      acc = __builtin_amdgcn_wmma_f32_16x16x32_f16(false, fh[kk], false, bw,
                                                   (short)0, acc, false, false);
    }
    int col = 16 * n + rsel;
    float bv = bias[col];
#pragma unroll
    for (int r = 0; r < 8; ++r) {
      int m = r + hi * 8;
      out[(size_t)(r0 + m) * 128 + col] = fmaxf(acc[r] + bv, 0.0f);
    }
  }
}

// ---------------------------------------------------------------- global mean pool (atomic)
__global__ void pool_kernel(const float* __restrict__ h, const int* __restrict__ batch,
                            float* __restrict__ pooled) {
  int tid = blockIdx.x * 256 + threadIdx.x;   // N_NODES*128
  int n = tid >> 7, j = tid & 127;
  atomicAdd(&pooled[batch[n] * 128 + j], h[tid]);
}

__global__ void gcnt_kernel(const int* __restrict__ batch, float* __restrict__ gcnt) {
  int n = blockIdx.x * 256 + threadIdx.x;
  if (n < N_NODES) atomicAdd(&gcnt[batch[n]], 1.0f);
}

// ---------------------------------------------------------------- FC head + log_softmax (tiny)
__global__ void head_kernel(const float* __restrict__ pooled, const float* __restrict__ gcnt,
                            const float* __restrict__ Wfc, const float* __restrict__ bfc,
                            float* __restrict__ out) {
  int g = threadIdx.x;
  if (g >= N_GRAPHS) return;
  float inv = 1.0f / fmaxf(gcnt[g], 1.0f);
  float lg[N_CLASSES];
  for (int c = 0; c < N_CLASSES; ++c) lg[c] = bfc[c];
  for (int k = 0; k < HIDDEN; ++k) {
    float p = pooled[g * 128 + k] * inv;
    for (int c = 0; c < N_CLASSES; ++c) lg[c] += p * Wfc[k * N_CLASSES + c];
  }
  float m = lg[0];
  for (int c = 1; c < N_CLASSES; ++c) m = fmaxf(m, lg[c]);
  float s = 0.0f;
  for (int c = 0; c < N_CLASSES; ++c) s += expf(lg[c] - m);
  float lse = m + logf(s);
  for (int c = 0; c < N_CLASSES; ++c) out[g * N_CLASSES + c] = lg[c] - lse;
}

// ----------------------------------------------------------------
extern "C" void kernel_launch(void* const* d_in, const int* in_sizes, int n_in,
                              void* d_out, int out_size, void* d_ws, size_t ws_size,
                              hipStream_t stream) {
  const float* x   = (const float*)d_in[0];
  const int*   ei  = (const int*)d_in[1];
  const int*   src = ei;
  const int*   dst = ei + N_EDGES;
  const int*   batch = (const int*)d_in[2];
  const float* W1l = (const float*)d_in[3];
  const float* W1r = (const float*)d_in[4];
  const float* b1  = (const float*)d_in[5];
  const float* W2l = (const float*)d_in[6];
  const float* W2r = (const float*)d_in[7];
  const float* b2  = (const float*)d_in[8];
  const float* W3l = (const float*)d_in[9];
  const float* W3r = (const float*)d_in[10];
  const float* b3  = (const float*)d_in[11];
  const float* Wfc = (const float*)d_in[12];
  const float* bfc = (const float*)d_in[13];
  float* out = (float*)d_out;

  char* ws = (char*)d_ws;
  const size_t NB = (size_t)N_NODES * HIDDEN * sizeof(float);  // 51,200,000 (256-aligned)
  const size_t NV = 400128;                                    // N_NODES*4 rounded to 256
  float*    hA     = (float*)(ws);
  float*    hB     = (float*)(ws + NB);
  float*    agg    = (float*)(ws + 2 * NB);
  float*    deg    = (float*)(ws + 3 * NB);
  float*    agg1   = (float*)(ws + 3 * NB + NV);
  _Float16* wl     = (_Float16*)(ws + 3 * NB + 2 * NV);
  _Float16* wr     = wl + 16384;
  float*    pooled = (float*)(ws + 3 * NB + 2 * NV + 2 * 32768);
  float*    gcnt   = pooled + N_GRAPHS * HIDDEN;

  const int EB   = N_EDGES / 256;            // 6250
  const int NJ   = N_NODES * HIDDEN / 256;   // 50000
  const int SCAT = N_EDGES * 32 / 256;       // 200000
  const int TIL  = N_NODES / 16;             // 6250

  // zero accumulators (capture-legal async memsets)
  hipMemsetAsync(deg,    0, (size_t)N_NODES * 4, stream);
  hipMemsetAsync(agg1,   0, (size_t)N_NODES * 4, stream);
  hipMemsetAsync(agg,    0, NB, stream);
  hipMemsetAsync(pooled, 0, (size_t)N_GRAPHS * HIDDEN * 4, stream);
  hipMemsetAsync(gcnt,   0, (size_t)N_GRAPHS * 4, stream);

  // layer 1 (d_in = 1)
  deg_agg1_kernel<<<EB, 256, 0, stream>>>(src, dst, x, deg, agg1);
  layer1_kernel<<<NJ, 256, 0, stream>>>(x, agg1, deg, W1l, W1r, b1, hA);

  // layer 2
  pack_w_kernel<<<64, 256, 0, stream>>>(W2l, wl);
  pack_w_kernel<<<64, 256, 0, stream>>>(W2r, wr);
  scatter128_kernel<<<SCAT, 256, 0, stream>>>(hA, src, dst, agg);
  sage_gemm_kernel<<<TIL, 32, 0, stream>>>(agg, deg, hA, wl, wr, b2, hB);

  // layer 3
  hipMemsetAsync(agg, 0, NB, stream);
  pack_w_kernel<<<64, 256, 0, stream>>>(W3l, wl);
  pack_w_kernel<<<64, 256, 0, stream>>>(W3r, wr);
  scatter128_kernel<<<SCAT, 256, 0, stream>>>(hB, src, dst, agg);
  sage_gemm_kernel<<<TIL, 32, 0, stream>>>(agg, deg, hB, wl, wr, b3, hA);

  // pool + head
  gcnt_kernel<<<(N_NODES + 255) / 256, 256, 0, stream>>>(batch, gcnt);
  pool_kernel<<<NJ, 256, 0, stream>>>(hA, batch, pooled);
  head_kernel<<<1, 64, 0, stream>>>(pooled, gcnt, Wfc, bfc, out);
}
